// VectorQuantizer_34737695490128
// MI455X (gfx1250) — matrix operand order repaired and checked by reading, not compiled, not run on hardware
//
#include <hip/hip_runtime.h>
#include <hip/hip_bf16.h>

#define K_CODES 512
#define DIM 64
#define ROWS_PER_BLOCK 256   // 8 waves x 32 rows (two 16-row tiles per wave)
#define THREADS 256
#define QPITCH 68            // LDS staging pitch (floats): 16B-aligned, de-conflicted

typedef __bf16 bf16_t;
typedef __attribute__((ext_vector_type(16))) __bf16 v16bf;
typedef __attribute__((ext_vector_type(8)))  float  v8f;

__device__ __forceinline__ unsigned bfbits(float f) {
  unsigned u = __builtin_bit_cast(unsigned, f);
  return (u + 0x7FFFu + ((u >> 16) & 1u)) >> 16;   // RNE bf16
}
__device__ __forceinline__ bf16_t bits2bf(unsigned short v) {
  return __builtin_bit_cast(bf16_t, v);
}
__device__ __forceinline__ bf16_t f2bf(float f) {
  return bits2bf((unsigned short)bfbits(f));
}
// monotone float -> uint map: a < b  <=>  sortkey(a) < sortkey(b)
__device__ __forceinline__ unsigned sortkey(float d) {
  unsigned u = __builtin_bit_cast(unsigned, d);
  return ((int)u < 0) ? ~u : (u | 0x80000000u);
}

struct P1 {
  unsigned code[K_CODES * (DIM / 2)];  // bf16 pairs, 64 KB
  float    norm[K_CODES];              // ||w||^2, 2 KB
};
union SU {
  P1    p1;
  float q[ROWS_PER_BLOCK * QPITCH];    // gathered rows staging, ~68 KB
};

// ---------------- init: zero workspace (counts[512] + loss accumulator) ----
__global__ void vq_init_kernel(unsigned* ws) {
  int t = blockIdx.x * blockDim.x + threadIdx.x;
  if (t < 1024) ws[t] = 0u;
}

// ---------------- main: distances via bf16 WMMA, argmin, gather, loss ------
__global__ void __launch_bounds__(THREADS)
vq_main_kernel(const float* __restrict__ x, const float* __restrict__ w,
               float* __restrict__ out, unsigned* __restrict__ counts,
               float* __restrict__ lossAcc, int nRows) {
  __shared__ SU    sU;
  __shared__ int   sIdx[ROWS_PER_BLOCK];
  __shared__ float sRed[8];

  const int t    = threadIdx.x;
  const int lane = t & 31;
  const int wave = t >> 5;
  const int hi   = lane >> 4;   // 0: lanes 0-15, 1: lanes 16-31
  const int col  = lane & 15;

  // ---- phase 0: codebook -> LDS (bf16) + ||w||^2 ----
  for (int c = t; c < K_CODES; c += THREADS) {
    const float4* wr = (const float4*)(w + c * DIM);
    float nrm = 0.f;
#pragma unroll
    for (int j = 0; j < DIM / 4; ++j) {
      float4 v = wr[j];
      nrm += v.x * v.x + v.y * v.y + v.z * v.z + v.w * v.w;
      sU.p1.code[c * (DIM / 2) + 2 * j]     = (bfbits(v.y) << 16) | bfbits(v.x);
      sU.p1.code[c * (DIM / 2) + 2 * j + 1] = (bfbits(v.w) << 16) | bfbits(v.z);
    }
    sU.p1.norm[c] = nrm;
  }
  __syncthreads();

  // ---- phase 1: per-wave 2x16 rows vs all 512 codes ----
  const int rowBase = blockIdx.x * ROWS_PER_BLOCK + wave * 32;

  // A fragments (16x32 bf16, ISA 7.12.2): lane holds row=col,
  // VGPR v <-> K = 16*(v>>2) + 2*(v&3) + 8*hi (+1 in high half)
  v16bf a00, a01, a10, a11;   // [tile][k-chunk]
  {
    const float* xr0 = x + (size_t)(rowBase + col) * DIM;
    const float* xr1 = x + (size_t)(rowBase + 16 + col) * DIM;
#pragma unroll
    for (int v = 0; v < 8; ++v) {
      int k0 = ((v >> 2) * 16) + ((v & 3) * 2) + hi * 8;
      float2 p0 = *(const float2*)(xr0 + k0);
      float2 p1 = *(const float2*)(xr0 + 32 + k0);
      float2 p2 = *(const float2*)(xr1 + k0);
      float2 p3 = *(const float2*)(xr1 + 32 + k0);
      a00[2 * v] = f2bf(p0.x);  a00[2 * v + 1] = f2bf(p0.y);
      a01[2 * v] = f2bf(p1.x);  a01[2 * v + 1] = f2bf(p1.y);
      a10[2 * v] = f2bf(p2.x);  a10[2 * v + 1] = f2bf(p2.y);
      a11[2 * v] = f2bf(p3.x);  a11[2 * v + 1] = f2bf(p3.y);
    }
  }

  float bestD[16];
  int   bestN[16];
#pragma unroll
  for (int r = 0; r < 16; ++r) { bestD[r] = 3.4e38f; bestN[r] = 0; }

  for (int cg = 0; cg < K_CODES / 16; ++cg) {
    const int n = cg * 16 + col;  // this lane's code column
    // B fragment (32x16 bf16): elems 0-7 K=8*hi+0..7, elems 8-15 K=16+8*hi+0..7
    const unsigned* cb = sU.p1.code + n * (DIM / 2) + hi * 4;
    uint4 qa = *(const uint4*)(cb);
    uint4 qb = *(const uint4*)(cb + 8);
    uint4 qc = *(const uint4*)(cb + 16);
    uint4 qd = *(const uint4*)(cb + 24);
    v16bf b0, b1;
#define PUT2(vec, j, p)                                            \
    (vec)[2 * (j)]     = bits2bf((unsigned short)((p) & 0xffffu)); \
    (vec)[2 * (j) + 1] = bits2bf((unsigned short)((p) >> 16));
    PUT2(b0, 0, qa.x) PUT2(b0, 1, qa.y) PUT2(b0, 2, qa.z) PUT2(b0, 3, qa.w)
    PUT2(b0, 4, qb.x) PUT2(b0, 5, qb.y) PUT2(b0, 6, qb.z) PUT2(b0, 7, qb.w)
    PUT2(b1, 0, qc.x) PUT2(b1, 1, qc.y) PUT2(b1, 2, qc.z) PUT2(b1, 3, qc.w)
    PUT2(b1, 4, qd.x) PUT2(b1, 5, qd.y) PUT2(b1, 6, qd.z) PUT2(b1, 7, qd.w)
#undef PUT2

    float wn = sU.p1.norm[n];
    // two independent accumulator chains -> fills WMMA hazard slots
    v8f acc0 = {}, acc1 = {};
    acc0 = __builtin_amdgcn_wmma_f32_16x16x32_bf16(false, a00, false, b0,
                                                   (short)0, acc0, false, false);
    acc1 = __builtin_amdgcn_wmma_f32_16x16x32_bf16(false, a10, false, b0,
                                                   (short)0, acc1, false, false);
    acc0 = __builtin_amdgcn_wmma_f32_16x16x32_bf16(false, a01, false, b1,
                                                   (short)0, acc0, false, false);
    acc1 = __builtin_amdgcn_wmma_f32_16x16x32_bf16(false, a11, false, b1,
                                                   (short)0, acc1, false, false);
    // D layout: VGPR r -> row (r + 8*hi) of the tile, lane -> column n
#pragma unroll
    for (int r = 0; r < 8; ++r) {
      float d0 = wn - 2.0f * acc0[r];
      if (d0 < bestD[r]) { bestD[r] = d0; bestN[r] = n; }
      float d1 = wn - 2.0f * acc1[r];
      if (d1 < bestD[8 + r]) { bestD[8 + r] = d1; bestN[8 + r] = n; }
    }
  }

  // branchless cross-lane argmin: lexicographic (dist, index) as one u64 key
#pragma unroll
  for (int r = 0; r < 16; ++r) {
    unsigned long long key =
        ((unsigned long long)sortkey(bestD[r]) << 32) | (unsigned)bestN[r];
#pragma unroll
    for (int m = 8; m >= 1; m >>= 1) {
      unsigned long long o = __shfl_xor(key, m, 32);
      key = (o < key) ? o : key;
    }
    if (col == 0)
      sIdx[wave * 32 + (r >> 3) * 16 + hi * 8 + (r & 7)] =
          (int)(unsigned)(key & 0xffffffffu);
  }
  __syncthreads();   // all waves done reading sU.p1 -> safe to reuse as sU.q

  // ---- phase 2: gather fp32 codebook row, loss partial, histogram ---------
  const int grow = blockIdx.x * ROWS_PER_BLOCK + t;   // one row per thread
  const int idx  = sIdx[t];
  const float4* wq = (const float4*)(w + (size_t)idx * DIM);
  const float4* xq = (const float4*)(x + (size_t)grow * DIM);
  float4* qrow = (float4*)(sU.q + t * QPITCH);        // 272B pitch: 16B aligned
  float s = 0.f;
#pragma unroll
  for (int j = 0; j < DIM / 4; ++j) {
    float4 wv = wq[j];
    float4 xv = xq[j];
    float dx = wv.x - xv.x, dy = wv.y - xv.y, dz = wv.z - xv.z, dw = wv.w - xv.w;
    s += dx * dx + dy * dy + dz * dz + dw * dw;
    qrow[j] = wv;
  }
  atomicAdd(&counts[idx], 1u);
  out[2 + (size_t)nRows * DIM + grow] = (float)idx;

#pragma unroll
  for (int m = 16; m >= 1; m >>= 1) s += __shfl_xor(s, m, 32);
  if (lane == 0) sRed[wave] = s;
  __syncthreads();

  // coalesced copy-out of quantized rows (d_out is offset by the loss scalar)
  float* outQ = out + 1 + (size_t)blockIdx.x * (ROWS_PER_BLOCK * DIM);
#pragma unroll
  for (int j = 0; j < (ROWS_PER_BLOCK * DIM) / THREADS; ++j) {
    int e = t + THREADS * j;                       // lane-consecutive -> coalesced
    outQ[e] = sU.q[(e >> 6) * QPITCH + (e & 63)];
  }
  if (t == 0) {
    float tot = 0.f;
#pragma unroll
    for (int i = 0; i < 8; ++i) tot += sRed[i];
    atomicAdd(lossAcc, tot);
  }
}

// ---------------- finalize: loss scalar + normalized perplexity ------------
__global__ void vq_fin_kernel(const unsigned* __restrict__ counts,
                              const float* __restrict__ lossAcc,
                              float* __restrict__ out, int nRows) {
  __shared__ float sh[256];
  int t = threadIdx.x;
  float h = 0.f;
  for (int c = t; c < K_CODES; c += 256) {
    float p = (float)counts[c] / (float)nRows;
    h -= p * logf(p + 1e-10f);
  }
  sh[t] = h;
  __syncthreads();
  for (int st = 128; st >= 1; st >>= 1) {
    if (t < st) sh[t] += sh[t + st];
    __syncthreads();
  }
  if (t == 0) {
    out[0] = 1.25f * lossAcc[0] / ((float)nRows * (float)DIM);   // q + 0.25*e
    out[1 + (size_t)nRows * DIM] = expf(sh[0]) / (float)K_CODES; // perplexity/512
  }
}

extern "C" void kernel_launch(void* const* d_in, const int* in_sizes, int n_in,
                              void* d_out, int out_size, void* d_ws, size_t ws_size,
                              hipStream_t stream) {
  const float* x = (const float*)d_in[0];   // [N, 64] fp32
  const float* w = (const float*)d_in[1];   // [512, 64] fp32
  float* out = (float*)d_out;
  const int nRows = in_sizes[0] / DIM;

  unsigned* counts  = (unsigned*)d_ws;                                    // 512 u32
  float*    lossAcc = (float*)((char*)d_ws + K_CODES * sizeof(unsigned)); // 1 f32

  vq_init_kernel<<<4, 256, 0, stream>>>((unsigned*)d_ws);
  vq_main_kernel<<<nRows / ROWS_PER_BLOCK, THREADS, 0, stream>>>(
      x, w, out, counts, lossAcc, nRows);
  vq_fin_kernel<<<1, 256, 0, stream>>>(counts, lossAcc, out, nRows);
}
